// QuantumMultiHeadAttention_73375221284954
// MI455X (gfx1250) — matrix operand order repaired and checked by reading, not compiled, not run on hardware
//
#include <hip/hip_runtime.h>

typedef __attribute__((ext_vector_type(16))) __bf16 v16bf;
typedef __attribute__((ext_vector_type(8)))  float  v8f;

#define DIMC   768
#define HEADS  12
#define DH     64
#define BATCH  4
#define SEQ    1024
#define ROWS   (BATCH*SEQ)   /* 4096 */
#define C3     (3*DIMC)      /* 2304 */

union Frag {
  v16bf v;
  int4  q[2];
};

__device__ __forceinline__ unsigned short f2bf(float f) {
  unsigned int u = __float_as_uint(f);
  return (unsigned short)((u + 0x7FFFu + ((u >> 16) & 1u)) >> 16);
}

__device__ __forceinline__ v8f zero8() {
  v8f z = {0.f,0.f,0.f,0.f,0.f,0.f,0.f,0.f};
  return z;
}

// A fragment (16x32 bf16): lane m = l%16; K chunks {k0..k0+7, k0+16..k0+23} (lanes<16)
//                          or {k0+8..k0+15, k0+24..k0+31} (lanes>=16). Row-major source.
__device__ __forceinline__ void load_a(Frag& f, const unsigned short* base, int ld,
                                       int row0, int k0, int lane) {
  int m   = lane & 15;
  int sel = (lane >> 4) & 1;
  const unsigned short* p = base + (size_t)(row0 + m) * ld + k0 + sel * 8;
  f.q[0] = *(const int4*)(p);
  f.q[1] = *(const int4*)(p + 16);
}

// B fragment (32x16 bf16): lane n = l%16; K chunks {k0+8h..+7, k0+16+8h..+7}, h = (l>=16).
// Source is row-major B^T (row n, contiguous k).
__device__ __forceinline__ void load_b(Frag& f, const unsigned short* base, int ld,
                                       int n0, int k0, int lane) {
  int n   = lane & 15;
  int sel = (lane >> 4) & 1;
  const unsigned short* p = base + (size_t)(n0 + n) * ld + k0 + sel * 8;
  f.q[0] = *(const int4*)(p);
  f.q[1] = *(const int4*)(p + 16);
}

__device__ __forceinline__ v8f wmma_bf(const Frag& a, const Frag& b, v8f c) {
  return __builtin_amdgcn_wmma_f32_16x16x32_bf16(false, a.v, false, b.v,
                                                 (short)0, c, false, false);
}

// ---------------- prep kernels ----------------

// W_qkv[r=a*768+c][cc=b2*256+dd] = sum_i A[i,a,b2]*S[i,c,dd]   (2304 x 768)
__global__ void build_wqkv(const float* __restrict__ A, const float* __restrict__ S,
                           unsigned short* __restrict__ W) {
  int idx = blockIdx.x * blockDim.x + threadIdx.x;
  if (idx >= C3 * DIMC) return;
  int r = idx / DIMC, cc = idx % DIMC;
  int a = r / 768, c = r % 768;
  int b2 = cc >> 8, dd = cc & 255;
  float s = 0.f;
  #pragma unroll
  for (int i = 0; i < 3; i++)
    s += A[i * 9 + a * 3 + b2] * S[(size_t)(i * 768 + c) * 256 + dd];
  W[idx] = f2bf(s);
}

// W_proj[r=a*256+c][cc=b2*256+dd] = sum_i A[i,a,b2]*S[i,c,dd]  (768 x 768)
__global__ void build_wproj(const float* __restrict__ A, const float* __restrict__ S,
                            unsigned short* __restrict__ W) {
  int idx = blockIdx.x * blockDim.x + threadIdx.x;
  if (idx >= DIMC * DIMC) return;
  int r = idx / DIMC, cc = idx % DIMC;
  int a = r >> 8, c = r & 255;
  int b2 = cc >> 8, dd = cc & 255;
  float s = 0.f;
  #pragma unroll
  for (int i = 0; i < 3; i++)
    s += A[i * 9 + a * 3 + b2] * S[(size_t)(i * 256 + c) * 256 + dd];
  W[idx] = f2bf(s);
}

// M = (I + E) @ W_sup * (1/sqrt(Dh))      (12x12, fp32)
__global__ void build_m(const float* __restrict__ Wsup, const float* __restrict__ E,
                        float* __restrict__ M) {
  int t = threadIdx.x;
  if (t < 144) {
    int i = t / 12, j = t % 12;
    float s = Wsup[i * 12 + j];
    for (int k = 0; k < 12; k++) s += E[i * 12 + k] * Wsup[k * 12 + j];
    M[t] = s * 0.125f;   // Dh=64 -> scale = 1/8
  }
}

__global__ void cvt_bf16(const float* __restrict__ src, unsigned short* __restrict__ dst, int n) {
  int i = blockIdx.x * blockDim.x + threadIdx.x;
  if (i < n) dst[i] = f2bf(src[i]);
}

// ---------------- QKV GEMM: 32x64 wave tile, double-buffered k-loop ----------------
__global__ __launch_bounds__(128) void gemm_qkv(
    const unsigned short* __restrict__ X, const unsigned short* __restrict__ W,
    const float* __restrict__ bias,
    unsigned short* __restrict__ qb, unsigned short* __restrict__ kb,
    unsigned short* __restrict__ vt) {
  int lane = threadIdx.x & 31;
  int wave = threadIdx.x >> 5;
  int job  = blockIdx.x * 4 + wave;      // 4608 jobs: 128 row-tiles(32) * 36 col-groups(64)
  int ntile = job / 36;
  int cg    = job % 36;
  int rowbase = ntile * 32;

  v8f acc[2][4];
  #pragma unroll
  for (int r = 0; r < 2; r++)
    #pragma unroll
    for (int t = 0; t < 4; t++) acc[r][t] = zero8();

  Frag a0[2], b0[4], a1[2], b1[4];

  auto loadset = [&](Frag* aa, Frag* bb, int k0) {
    load_a(aa[0], X, DIMC, rowbase,      k0, lane);
    load_a(aa[1], X, DIMC, rowbase + 16, k0, lane);
    #pragma unroll
    for (int t = 0; t < 4; t++)
      load_b(bb[t], W, DIMC, cg * 64 + t * 16, k0, lane);
  };
  auto compute = [&](Frag* aa, Frag* bb) {
    #pragma unroll
    for (int t = 0; t < 4; t++) acc[0][t] = wmma_bf(aa[0], bb[t], acc[0][t]);
    #pragma unroll
    for (int t = 0; t < 4; t++) acc[1][t] = wmma_bf(aa[1], bb[t], acc[1][t]);
  };

  loadset(a0, b0, 0);
  for (int k0 = 0; k0 < DIMC; k0 += 64) {
    loadset(a1, b1, k0 + 32);            // in flight while set0 computes
    compute(a0, b0);
    if (k0 + 64 < DIMC) loadset(a0, b0, k0 + 64);
    compute(a1, b1);
  }

  int nn  = lane & 15;
  int sel = (lane >> 4) & 1;
  int bidx = rowbase >> 10;        // batch (32-aligned rows never straddle)

  #pragma unroll
  for (int r = 0; r < 2; r++) {
    int nloc = (rowbase & 1023) + r * 16;
    #pragma unroll
    for (int t = 0; t < 4; t++) {
      int f  = cg * 64 + t * 16 + nn;
      float bv = bias[f];
      if (f < DIMC) {                                   // Q: (B,H,N,Dh)
        int h = f >> 6, d = f & 63;
        unsigned short* dst = qb + ((size_t)(bidx * HEADS + h) * SEQ) * DH + d;
        #pragma unroll
        for (int i = 0; i < 8; i++) {
          int n = nloc + sel * 8 + i;
          dst[(size_t)n * DH] = f2bf(acc[r][t][i] + bv);
        }
      } else if (f < 2 * DIMC) {                        // K: (B,H,N,Dh)
        int f2 = f - DIMC;
        int h = f2 >> 6, d = f2 & 63;
        unsigned short* dst = kb + ((size_t)(bidx * HEADS + h) * SEQ) * DH + d;
        #pragma unroll
        for (int i = 0; i < 8; i++) {
          int n = nloc + sel * 8 + i;
          dst[(size_t)n * DH] = f2bf(acc[r][t][i] + bv);
        }
      } else {                                          // V transposed: (B,H,Dh,N)
        int f2 = f - 2 * DIMC;
        int h = f2 >> 6, d = f2 & 63;
        unsigned short* dst = vt + ((size_t)(bidx * HEADS + h) * DH + d) * SEQ;
        #pragma unroll
        for (int i = 0; i < 8; i++) {
          int n = nloc + sel * 8 + i;
          dst[n] = f2bf(acc[r][t][i] + bv);
        }
      }
    }
  }
}

// ---------------- fused attention ----------------
__device__ __forceinline__ float mixdot(const float* sp, const float* m) {
  const float4* s4 = (const float4*)sp;
  float4 a = s4[0], b = s4[1], c = s4[2];
  return a.x*m[0] + a.y*m[1] + a.z*m[2] + a.w*m[3]
       + b.x*m[4] + b.y*m[5] + b.z*m[6] + b.w*m[7]
       + c.x*m[8] + c.y*m[9] + c.z*m[10] + c.w*m[11];
}

__global__ __launch_bounds__(384) void attn_fused(
    const unsigned short* __restrict__ qb, const unsigned short* __restrict__ kb,
    const unsigned short* __restrict__ vt, const float* __restrict__ M,
    unsigned short* __restrict__ ob) {
  __shared__ __align__(16) float          s_mix[512][12];          // [pos(16x32)][head]
  __shared__ __align__(16) unsigned short s_p[HEADS][16][32];      // P, row-major per head

  int lane = threadIdx.x & 31;
  int h    = threadIdx.x >> 5;           // head = wave id (0..11)
  int bidx = blockIdx.x >> 6;
  int ntile = blockIdx.x & 63;
  int nbase = ntile * 16;
  int nn  = lane & 15;
  int sel = (lane >> 4) & 1;

  float mrow[12];
  #pragma unroll
  for (int j = 0; j < 12; j++) mrow[j] = M[h * 12 + j];

  const unsigned short* qh = qb + (size_t)(bidx * HEADS + h) * SEQ * DH;
  const unsigned short* kh = kb + (size_t)(bidx * HEADS + h) * SEQ * DH;
  const unsigned short* vh = vt + (size_t)(bidx * HEADS + h) * DH * SEQ;

  Frag qa, qc;
  load_a(qa, qh, DH, nbase, 0, lane);
  load_a(qc, qh, DH, nbase, 32, lane);

  v8f o[4];
  #pragma unroll
  for (int t = 0; t < 4; t++) o[t] = zero8();
  float rm[8], rl[8];
  #pragma unroll
  for (int i = 0; i < 8; i++) { rm[i] = -1e30f; rl[i] = 0.f; }

  // pull first chunk's K/V toward the caches
  __builtin_prefetch(kh + (size_t)lane * DH, 0, 1);
  __builtin_prefetch(vh + (size_t)lane * SEQ, 0, 1);
  __builtin_prefetch(vh + (size_t)(lane + 32) * SEQ, 0, 1);

  for (int mt = 0; mt < SEQ; mt += 32) {
    // ---- K fragments for both column tiles, one clause ----
    Frag kf0, kf1, kf2, kf3;
    load_b(kf0, kh, DH, mt,       0, lane);
    load_b(kf1, kh, DH, mt,      32, lane);
    load_b(kf2, kh, DH, mt + 16,  0, lane);
    load_b(kf3, kh, DH, mt + 16, 32, lane);

    v8f s0 = zero8(), s1 = zero8();
    s0 = wmma_bf(qa, kf0, s0);
    s0 = wmma_bf(qc, kf1, s0);
    s1 = wmma_bf(qa, kf2, s1);
    s1 = wmma_bf(qc, kf3, s1);

    // ---- V fragments for THIS chunk: latency hides under mixing/softmax ----
    Frag vfr[4];
    load_b(vfr[0], vh, SEQ,  0, mt, lane);
    load_b(vfr[1], vh, SEQ, 16, mt, lane);
    load_b(vfr[2], vh, SEQ, 32, mt, lane);
    load_b(vfr[3], vh, SEQ, 48, mt, lane);

    // ---- prefetch next chunk's K/V rows ----
    if (mt + 32 < SEQ) {
      __builtin_prefetch(kh + (size_t)(mt + 32 + lane) * DH, 0, 1);
      __builtin_prefetch(vh + (size_t)lane * SEQ + mt + 32, 0, 1);
      __builtin_prefetch(vh + (size_t)(lane + 32) * SEQ + mt + 32, 0, 1);
    }

    // ---- stage raw scores for cross-head mixing ----
    #pragma unroll
    for (int i = 0; i < 8; i++) {
      int row = i + sel * 8;
      s_mix[row * 32 + nn][h]      = s0[i];
      s_mix[row * 32 + 16 + nn][h] = s1[i];
    }
    __syncthreads();

    // ---- ent = tanh(M @ scores) ----
    float e0[8], e1[8];
    #pragma unroll
    for (int i = 0; i < 8; i++) {
      int row = i + sel * 8;
      e0[i] = tanhf(mixdot(&s_mix[row * 32 + nn][0], mrow));
      e1[i] = tanhf(mixdot(&s_mix[row * 32 + 16 + nn][0], mrow));
    }
    __syncthreads();   // s_mix free for next chunk

    // ---- online softmax update (rows live across 16-lane halves) ----
    #pragma unroll
    for (int i = 0; i < 8; i++) {
      float tmax = fmaxf(e0[i], e1[i]);
      #pragma unroll
      for (int m2 = 8; m2 >= 1; m2 >>= 1) tmax = fmaxf(tmax, __shfl_xor(tmax, m2, 32));
      float mn   = fmaxf(rm[i], tmax);
      float corr = __expf(rm[i] - mn);
      float p0 = __expf(e0[i] - mn);
      float p1 = __expf(e1[i] - mn);
      float ps = p0 + p1;
      #pragma unroll
      for (int m2 = 8; m2 >= 1; m2 >>= 1) ps += __shfl_xor(ps, m2, 32);
      rl[i] = rl[i] * corr + ps;
      rm[i] = mn;
      #pragma unroll
      for (int t = 0; t < 4; t++) o[t][i] *= corr;
      int row = i + sel * 8;
      s_p[h][row][nn]      = f2bf(p0);
      s_p[h][row][16 + nn] = f2bf(p1);
    }
    __syncthreads();

    // ---- O += P (16x32) @ V (32x64), V fragments already resident ----
    Frag pf;
    {
      const unsigned short* p = &s_p[h][0][0] + nn * 32 + sel * 8;
      pf.q[0] = *(const int4*)(p);
      pf.q[1] = *(const int4*)(p + 16);
    }
    #pragma unroll
    for (int t = 0; t < 4; t++) o[t] = wmma_bf(pf, vfr[t], o[t]);
  }

  // ---- normalize + store to (B,N,C) bf16 ----
  unsigned short* dst = ob + ((size_t)(bidx * SEQ + nbase)) * DIMC + h * DH;
  #pragma unroll
  for (int t = 0; t < 4; t++) {
    #pragma unroll
    for (int i = 0; i < 8; i++) {
      int row = i + sel * 8;
      dst[(size_t)row * DIMC + t * 16 + nn] = f2bf(o[t][i] / rl[i]);
    }
  }
}

// ---------------- projection GEMM: 32x64 wave tile, double-buffered ----------------
__global__ __launch_bounds__(128) void gemm_proj(
    const unsigned short* __restrict__ X, const unsigned short* __restrict__ W,
    const float* __restrict__ bias, float* __restrict__ out) {
  int lane = threadIdx.x & 31;
  int wave = threadIdx.x >> 5;
  int job  = blockIdx.x * 4 + wave;      // 1536 jobs: 128 row-tiles(32) * 12 col-groups(64)
  int ntile = job / 12;
  int cg    = job % 12;
  int rowbase = ntile * 32;

  v8f acc[2][4];
  #pragma unroll
  for (int r = 0; r < 2; r++)
    #pragma unroll
    for (int t = 0; t < 4; t++) acc[r][t] = zero8();

  Frag a0[2], b0[4], a1[2], b1[4];

  auto loadset = [&](Frag* aa, Frag* bb, int k0) {
    load_a(aa[0], X, DIMC, rowbase,      k0, lane);
    load_a(aa[1], X, DIMC, rowbase + 16, k0, lane);
    #pragma unroll
    for (int t = 0; t < 4; t++)
      load_b(bb[t], W, DIMC, cg * 64 + t * 16, k0, lane);
  };
  auto compute = [&](Frag* aa, Frag* bb) {
    #pragma unroll
    for (int t = 0; t < 4; t++) acc[0][t] = wmma_bf(aa[0], bb[t], acc[0][t]);
    #pragma unroll
    for (int t = 0; t < 4; t++) acc[1][t] = wmma_bf(aa[1], bb[t], acc[1][t]);
  };

  loadset(a0, b0, 0);
  for (int k0 = 0; k0 < DIMC; k0 += 64) {
    loadset(a1, b1, k0 + 32);
    compute(a0, b0);
    if (k0 + 64 < DIMC) loadset(a0, b0, k0 + 64);
    compute(a1, b1);
  }

  int nn  = lane & 15;
  int sel = (lane >> 4) & 1;
  #pragma unroll
  for (int r = 0; r < 2; r++) {
    #pragma unroll
    for (int t = 0; t < 4; t++) {
      int f = cg * 64 + t * 16 + nn;
      float bv = bias[f];
      #pragma unroll
      for (int i = 0; i < 8; i++) {
        int row = rowbase + r * 16 + i + sel * 8;
        out[(size_t)row * DIMC + f] = acc[r][t][i] + bv;
      }
    }
  }
}

// ---------------- launcher ----------------
extern "C" void kernel_launch(void* const* d_in, const int* in_sizes, int n_in,
                              void* d_out, int out_size, void* d_ws, size_t ws_size,
                              hipStream_t stream) {
  const float* x   = (const float*)d_in[0];
  const float* Aq  = (const float*)d_in[1];
  const float* Sq  = (const float*)d_in[2];
  const float* bq  = (const float*)d_in[3];
  const float* Ap  = (const float*)d_in[4];
  const float* Sp  = (const float*)d_in[5];
  const float* bp  = (const float*)d_in[6];
  const float* Wsp = (const float*)d_in[7];
  const float* Ee  = (const float*)d_in[8];

  char* ws = (char*)d_ws;
  size_t off = 0;
  auto carve = [&](size_t bytes) -> void* {
    void* p = ws + off;
    off += (bytes + 255) & ~(size_t)255;
    return p;
  };
  unsigned short* wqkv  = (unsigned short*)carve((size_t)C3 * DIMC * 2);
  unsigned short* wproj = (unsigned short*)carve((size_t)DIMC * DIMC * 2);
  float*          Mmix  = (float*)carve(144 * 4);
  unsigned short* xbf   = (unsigned short*)carve((size_t)ROWS * DIMC * 2);
  unsigned short* qbuf  = (unsigned short*)carve((size_t)BATCH * HEADS * SEQ * DH * 2);
  unsigned short* kbuf  = (unsigned short*)carve((size_t)BATCH * HEADS * SEQ * DH * 2);
  unsigned short* vtb   = (unsigned short*)carve((size_t)BATCH * HEADS * DH * SEQ * 2);
  unsigned short* obuf  = (unsigned short*)carve((size_t)ROWS * DIMC * 2);

  build_wqkv <<<(C3 * DIMC + 255) / 256, 256, 0, stream>>>(Aq, Sq, wqkv);
  build_wproj<<<(DIMC * DIMC + 255) / 256, 256, 0, stream>>>(Ap, Sp, wproj);
  build_m    <<<1, 192, 0, stream>>>(Wsp, Ee, Mmix);
  cvt_bf16   <<<(ROWS * DIMC + 255) / 256, 256, 0, stream>>>(x, xbf, ROWS * DIMC);

  gemm_qkv   <<<(ROWS / 32) * 36 / 4, 128, 0, stream>>>(xbf, wqkv, bq, qbuf, kbuf, vtb);
  attn_fused <<<BATCH * (SEQ / 16), 384, 0, stream>>>(qbuf, kbuf, vtb, Mmix, obuf);
  gemm_proj  <<<(ROWS / 32) * 12 / 4, 128, 0, stream>>>(obuf, wproj, bp, (float*)d_out);
}
